// Model_36953898615064
// MI455X (gfx1250) — matrix operand order repaired and checked by reading, not compiled, not run on hardware
//
#include <hip/hip_runtime.h>
#include <math.h>

#define U_CNT 60000
#define I_CNT 40000
#define N_CNT 100000
#define DIM 64
#define LAYERS 2
#define NNZ_TOT 500000
#define NNZ_HALF_C 250000
#define WLK 4
#define BATCH 4096
#define TMP_C 0.2f
#define SSL_TEMP_C 0.2f
#define SSL_REG_C 0.1f
#define SPARSE_REG_C 0.01f
#define BIAS_C 0.0001f

typedef __attribute__((ext_vector_type(16))) __bf16 v16bf;
typedef __attribute__((ext_vector_type(8)))  float  v8f;

__device__ __forceinline__ float gate_fn(float logit, float u) {
    float ep = (2.0f * BIAS_C - 1.0f) * u + (1.0f - BIAS_C);
    float noise = logf(ep) - log1pf(-ep);
    return 1.0f / (1.0f + expf(-(noise + logit) * (1.0f / TMP_C)));
}

// Fragment-layout helpers (v_wmma_f32_16x16x32_bf16):
//  A (16x32): lane = half*16 + m ; elem j -> K = kc*32 + half*16 + j
//  B (32x16): lane = half*16 + n ; elem j -> K = kc*32 + half*16 + j
//  C (16x16): lane<16 -> N=lane rows 0..7 ; lane>=16 -> rows 8..15
__device__ __forceinline__ void a_swizzle(int k, int& kc, int& half, int& j) {
    kc = k >> 5;
    int w = k & 31;
    half = (w >> 4) & 1;
    j = w & 15;
}

// ---------------- elementwise / graph kernels (f32, memory-bound) ----------

__global__ void concat_kernel(const float* __restrict__ ue,
                              const float* __restrict__ ie,
                              float* __restrict__ e0) {
    int idx = blockIdx.x * blockDim.x + threadIdx.x;
    if (idx >= N_CNT * DIM) return;
    int n = idx >> 6, d = idx & 63;
    e0[idx] = (n < U_CNT) ? ue[n * DIM + d] : ie[(n - U_CNT) * DIM + d];
}

__global__ void accum_kernel(float* __restrict__ dst, const float* __restrict__ src) {
    int idx = blockIdx.x * blockDim.x + threadIdx.x;
    if (idx >= N_CNT * DIM) return;
    dst[idx] += src[idx];
}

__global__ void spmm_kernel(float* __restrict__ out, const float* __restrict__ x,
                            const float* __restrict__ val, const float* __restrict__ mask,
                            const int* __restrict__ row, const int* __restrict__ col) {
    int idx = blockIdx.x * blockDim.x + threadIdx.x;
    if (idx >= NNZ_TOT * DIM) return;
    int e = idx >> 6, d = idx & 63;
    float v = val[e];
    if (mask) v *= mask[e];
    unsafeAtomicAdd(&out[row[e] * DIM + d], v * x[col[e] * DIM + d]);
}

__global__ void mix_kernel(float* __restrict__ out, const float* __restrict__ e,
                           const int* __restrict__ walk, const float* __restrict__ nm) {
    int idx = blockIdx.x * blockDim.x + threadIdx.x;
    if (idx >= N_CNT * DIM) return;
    int n = idx >> 6, d = idx & 63;
    const int* w = walk + n * WLK;
    float mp = 0.25f * (e[w[0] * DIM + d] + e[w[1] * DIM + d] +
                        e[w[2] * DIM + d] + e[w[3] * DIM + d]);
    float mv = nm[n];
    out[idx] = mv * e[idx] + (1.0f - mv) * mp;
}

__global__ void reduce_mul_sum_kernel(const float* __restrict__ a, const float* __restrict__ b,
                                      int n, float* __restrict__ target, float scale) {
    __shared__ float sdata[256];
    float s = 0.0f;
    for (int i = blockIdx.x * blockDim.x + threadIdx.x; i < n; i += gridDim.x * blockDim.x)
        s += b ? a[i] * b[i] : a[i];
    sdata[threadIdx.x] = s;
    __syncthreads();
    for (int o = 128; o > 0; o >>= 1) {
        if (threadIdx.x < o) sdata[threadIdx.x] += sdata[threadIdx.x + o];
        __syncthreads();
    }
    if (threadIdx.x == 0) unsafeAtomicAdd(target, sdata[0] * scale);
}

__global__ void gather_kernel(const float* __restrict__ vsum,
                              const int* __restrict__ uid, const int* __restrict__ pid,
                              const int* __restrict__ nid,
                              float* __restrict__ gu, float* __restrict__ gp,
                              float* __restrict__ gn) {
    int idx = blockIdx.x * blockDim.x + threadIdx.x;
    if (idx >= BATCH * DIM) return;
    int b = idx >> 6, d = idx & 63;
    const float s = 1.0f / 3.0f;  // mean over L+1 = 3 embeddings
    gu[idx] = vsum[uid[b] * DIM + d] * s;
    gp[idx] = vsum[(U_CNT + pid[b]) * DIM + d] * s;
    gn[idx] = vsum[(U_CNT + nid[b]) * DIM + d] * s;
}

__global__ void bpr_kernel(const float* __restrict__ gu, const float* __restrict__ gp,
                           const float* __restrict__ gn, float* __restrict__ target) {
    int b = blockIdx.x * blockDim.x + threadIdx.x;
    if (b >= BATCH) return;
    float dp = 0.0f, dn = 0.0f;
    for (int d = 0; d < DIM; ++d) {
        float u = gu[b * DIM + d];
        dp += u * gp[b * DIM + d];
        dn += u * gn[b * DIM + d];
    }
    float x = -(dp - dn);                       // softplus(-diff) = -log_sigmoid(diff)
    float l = (x > 20.0f) ? x : log1pf(expf(x));
    unsafeAtomicAdd(target, l * (1.0f / BATCH));
}

__global__ void normalize_kernel(float* __restrict__ x) {
    int r = blockIdx.x * blockDim.x + threadIdx.x;
    if (r >= BATCH) return;
    float s = 0.0f;
    for (int d = 0; d < DIM; ++d) { float v = x[r * DIM + d]; s += v * v; }
    float inv = 1.0f / fmaxf(sqrtf(s), 1e-12f);
    for (int d = 0; d < DIM; ++d) x[r * DIM + d] *= inv;
}

// f32 row-major (BATCH x 64) -> bf16 A-fragment layout [tile][lane][kc*16+j]
__global__ void frag_a_convert(const float* __restrict__ x, __bf16* __restrict__ out) {
    int idx = blockIdx.x * blockDim.x + threadIdx.x;
    if (idx >= (BATCH / 16) * 32 * 32) return;
    int j = idx & 31;
    int lane = (idx >> 5) & 31;
    int tile = idx >> 10;
    int kc = j >> 4, jj = j & 15;
    int m = lane & 15, half = lane >> 4;
    int row = tile * 16 + m;
    int k = kc * 32 + half * 16 + jj;
    out[idx] = (__bf16)x[row * DIM + k];
}

// f32 row-major (BATCH x 64) -> bf16 B-fragment layout (transposed use: B = x^T)
__global__ void frag_b_convert(const float* __restrict__ x, __bf16* __restrict__ out) {
    int idx = blockIdx.x * blockDim.x + threadIdx.x;
    if (idx >= (BATCH / 16) * 32 * 32) return;
    int j = idx & 31;
    int lane = (idx >> 5) & 31;
    int tile = idx >> 10;
    int kc = j >> 4, jj = j & 15;
    int n = lane & 15, half = lane >> 4;
    int col = tile * 16 + n;
    int k = kc * 32 + half * 16 + jj;
    out[idx] = (__bf16)x[col * DIM + k];
}

// ---------------- WMMA kernels (bf16 in, f32 accumulate) -------------------

__global__ void edge_mlp_kernel(const float* __restrict__ e,
                                const float* __restrict__ w1,   // 128x64
                                const float* __restrict__ b1,   // 64
                                const float* __restrict__ w2,   // 64
                                const float* __restrict__ b2p,  // 1
                                const float* __restrict__ eps,  // NNZ
                                const int* __restrict__ row, const int* __restrict__ col,
                                float* __restrict__ maskOut) {
    // W1 staged fragment-major: w1t[kc][c][khalf][n][j] = w1[kc*32+khalf*16+j][c*16+n]
    __shared__ __align__(32) __bf16 w1t[4][4][2][16][16];
    __shared__ float b1s[64], w2s[64];
    // cat staged fragment-major: catt[wave][kc][lane][j]
    __shared__ __align__(32) __bf16 catt[4][4][32][16];
    __shared__ float red[128][8];
    const int tid = threadIdx.x, wid = tid >> 5, lane = tid & 31;
    for (int t = tid; t < 128 * 64; t += 128) {
        int k = t >> 6, cidx = t & 63;
        int kc = k >> 5, wk = k & 31;
        w1t[kc][cidx >> 4][wk >> 4][cidx & 15][wk & 15] = (__bf16)w1[t];
    }
    if (tid < 64) { b1s[tid] = b1[tid]; w2s[tid] = w2[tid]; }
    __syncthreads();
    const int tile = blockIdx.x * 4 + wid;
    const bool active = tile < (NNZ_TOT / 16);
    if (active) {
        for (int t = lane; t < 16 * 128; t += 32) {
            int r = t >> 7, k = t & 127;
            int eidx = tile * 16 + r;
            float v = (k < 64) ? e[row[eidx] * DIM + k] : e[col[eidx] * DIM + (k - 64)];
            int kc, half, j;
            a_swizzle(k, kc, half, j);
            catt[wid][kc][half * 16 + r][j] = (__bf16)v;
        }
    }
    __syncthreads();
    if (active) {
        const int n = lane & 15, khalf = lane >> 4;
        v16bf aF[4];
#pragma unroll
        for (int kc = 0; kc < 4; ++kc)
            aF[kc] = *(const v16bf*)&catt[wid][kc][lane][0];
        float pl[8];
#pragma unroll
        for (int r = 0; r < 8; ++r) pl[r] = 0.0f;
#pragma unroll
        for (int c = 0; c < 4; ++c) {
            float b1v = b1s[c * 16 + n];
            float w2v = w2s[c * 16 + n];
            v8f acc = {};
#pragma unroll
            for (int kc = 0; kc < 4; ++kc) {
                v16bf b = *(const v16bf*)&w1t[kc][c][khalf][n][0];
                acc = __builtin_amdgcn_wmma_f32_16x16x32_bf16(false, aF[kc], false, b,
                                                              (short)0, acc, false, false);
            }
#pragma unroll
            for (int r = 0; r < 8; ++r) {
                float h = fmaxf(acc[r] + b1v, 0.0f);
                pl[r] += h * w2v;
            }
        }
#pragma unroll
        for (int r = 0; r < 8; ++r) red[tid][r] = pl[r];
    }
    __syncthreads();
    if (active && lane < 16) {
        int base = wid * 32 + ((lane < 8) ? 0 : 16);
        int rr = lane & 7;
        float logit = b2p[0];
#pragma unroll
        for (int l = 0; l < 16; ++l) logit += red[base + l][rr];
        int eidx = tile * 16 + lane;
        maskOut[eidx] = gate_fn(logit, eps[eidx]);
    }
}

__global__ void node_mlp_kernel(const float* __restrict__ e,
                                const float* __restrict__ w1,   // 64x64
                                const float* __restrict__ b1, const float* __restrict__ w2,
                                const float* __restrict__ b2p, const float* __restrict__ eps,
                                float* __restrict__ maskOut) {
    __shared__ __align__(32) __bf16 w1t[2][4][2][16][16];
    __shared__ float b1s[64], w2s[64];
    __shared__ __align__(32) __bf16 catt[4][2][32][16];
    __shared__ float red[128][8];
    const int tid = threadIdx.x, wid = tid >> 5, lane = tid & 31;
    for (int t = tid; t < 64 * 64; t += 128) {
        int k = t >> 6, cidx = t & 63;
        int kc = k >> 5, wk = k & 31;
        w1t[kc][cidx >> 4][wk >> 4][cidx & 15][wk & 15] = (__bf16)w1[t];
    }
    if (tid < 64) { b1s[tid] = b1[tid]; w2s[tid] = w2[tid]; }
    __syncthreads();
    const int tile = blockIdx.x * 4 + wid;
    const bool active = tile < (N_CNT / 16);
    if (active) {
        for (int t = lane; t < 16 * 64; t += 32) {
            int r = t >> 6, k = t & 63;
            float v = e[(tile * 16 + r) * DIM + k];
            int kc, half, j;
            a_swizzle(k, kc, half, j);
            catt[wid][kc][half * 16 + r][j] = (__bf16)v;
        }
    }
    __syncthreads();
    if (active) {
        const int n = lane & 15, khalf = lane >> 4;
        v16bf aF[2];
#pragma unroll
        for (int kc = 0; kc < 2; ++kc)
            aF[kc] = *(const v16bf*)&catt[wid][kc][lane][0];
        float pl[8];
#pragma unroll
        for (int r = 0; r < 8; ++r) pl[r] = 0.0f;
#pragma unroll
        for (int c = 0; c < 4; ++c) {
            float b1v = b1s[c * 16 + n];
            float w2v = w2s[c * 16 + n];
            v8f acc = {};
#pragma unroll
            for (int kc = 0; kc < 2; ++kc) {
                v16bf b = *(const v16bf*)&w1t[kc][c][khalf][n][0];
                acc = __builtin_amdgcn_wmma_f32_16x16x32_bf16(false, aF[kc], false, b,
                                                              (short)0, acc, false, false);
            }
#pragma unroll
            for (int r = 0; r < 8; ++r) {
                float h = fmaxf(acc[r] + b1v, 0.0f);
                pl[r] += h * w2v;
            }
        }
#pragma unroll
        for (int r = 0; r < 8; ++r) red[tid][r] = pl[r];
    }
    __syncthreads();
    if (active && lane < 16) {
        int base = wid * 32 + ((lane < 8) ? 0 : 16);
        int rr = lane & 7;
        float logit = b2p[0];
#pragma unroll
        for (int l = 0; l < 16; ++l) logit += red[base + l][rr];
        int nidx = tile * 16 + lane;
        maskOut[nidx] = gate_fn(logit, eps[nidx]);
    }
}

// mean over rows of (a_i . b_i)/T - log(sum_j exp((a_i . b_j)/T)) -> atomic add
__global__ void ssl_kernel(const __bf16* __restrict__ afrag, const __bf16* __restrict__ bfrag,
                           const float* __restrict__ a32, const float* __restrict__ b32,
                           float* __restrict__ target) {
    __shared__ float red[128][8];
    const int tid = threadIdx.x, wid = tid >> 5, lane = tid & 31;
    const int tile = blockIdx.x * 4 + wid;
    const __bf16* ap = afrag + ((size_t)tile * 32 + lane) * 32;
    v16bf a0 = *(const v16bf*)ap;
    v16bf a1 = *(const v16bf*)(ap + 16);
    float se[8];
#pragma unroll
    for (int r = 0; r < 8; ++r) se[r] = 0.0f;
    const float invT = 1.0f / SSL_TEMP_C;
    for (int ct = 0; ct < BATCH / 16; ++ct) {
        const __bf16* bp = bfrag + ((size_t)ct * 32 + lane) * 32;
        v16bf b0 = *(const v16bf*)bp;
        v16bf b1 = *(const v16bf*)(bp + 16);
        v8f acc = {};
        acc = __builtin_amdgcn_wmma_f32_16x16x32_bf16(false, a0, false, b0, (short)0, acc, false, false);
        acc = __builtin_amdgcn_wmma_f32_16x16x32_bf16(false, a1, false, b1, (short)0, acc, false, false);
#pragma unroll
        for (int r = 0; r < 8; ++r) se[r] += expf(acc[r] * invT);  // |score| <= 5, safe
    }
#pragma unroll
    for (int r = 0; r < 8; ++r) red[tid][r] = se[r];
    __syncthreads();
    if (lane < 16) {
        int base = wid * 32 + ((lane < 8) ? 0 : 16);
        float s = 0.0f;
#pragma unroll
        for (int l = 0; l < 16; ++l) s += red[base + l][lane & 7];
        int rg = tile * 16 + lane;
        float pos = 0.0f;
        for (int d = 0; d < DIM; ++d) pos += a32[rg * DIM + d] * b32[rg * DIM + d];
        float res = pos * invT - logf(s);
        unsafeAtomicAdd(target, res * (1.0f / BATCH));
    }
}

__global__ void finalize_kernel(const float* __restrict__ scal, float* __restrict__ out) {
    out[0] = scal[0] + scal[1] + scal[2] - SSL_REG_C * scal[3] +
             SPARSE_REG_C * (scal[4] + scal[5]);
}

// ---------------------------------------------------------------------------

extern "C" void kernel_launch(void* const* d_in, const int* in_sizes, int n_in,
                              void* d_out, int out_size, void* d_ws, size_t ws_size,
                              hipStream_t stream) {
    (void)in_sizes; (void)n_in; (void)out_size; (void)ws_size;
    const float* user_emb = (const float*)d_in[0];
    const float* item_emb = (const float*)d_in[1];
    const float* edge_w1  = (const float*)d_in[2];
    const float* edge_b1  = (const float*)d_in[3];
    const float* edge_w2  = (const float*)d_in[4];
    const float* edge_b2  = (const float*)d_in[5];
    const float* node_w1  = (const float*)d_in[6];
    const float* node_b1  = (const float*)d_in[7];
    const float* node_w2  = (const float*)d_in[8];
    const float* node_b2  = (const float*)d_in[9];
    const float* adj_val  = (const float*)d_in[10];
    const float* edge_eps = (const float*)d_in[11];
    const float* node_eps = (const float*)d_in[12];
    const int*   adj_row  = (const int*)d_in[13];
    const int*   adj_col  = (const int*)d_in[14];
    const int*   walk     = (const int*)d_in[15];
    const int*   user_id  = (const int*)d_in[16];
    const int*   pos_item = (const int*)d_in[17];
    const int*   neg_item = (const int*)d_in[18];

    const size_t ND = (size_t)N_CNT * DIM;
    const size_t BD = (size_t)BATCH * DIM;
    float* ws    = (float*)d_ws;
    float* e0    = ws;
    float* eA    = e0 + ND;
    float* eB    = eA + ND;
    float* vsum  = eB + ND;
    float* emask = vsum + ND;                          // LAYERS * NNZ
    float* nmask = emask + (size_t)LAYERS * NNZ_TOT;   // LAYERS * N
    float* gbuf  = nmask + (size_t)LAYERS * N_CNT;     // 9 * BATCH * DIM
    float* scal  = gbuf + 9 * BD;                      // 16 scalars
    __bf16* fr   = (__bf16*)(scal + 16);               // 6 * BD bf16 fragment buffers
    __bf16* fr_bu = fr + 0 * BD;
    __bf16* fr_bi = fr + 1 * BD;
    __bf16* fr_eu = fr + 2 * BD;
    __bf16* fr_nu = fr + 3 * BD;
    __bf16* fr_ei = fr + 4 * BD;
    __bf16* fr_ni = fr + 5 * BD;

    const int ND_BLOCKS    = (int)((ND + 255) / 256);
    const int SPMM_BLOCKS  = (int)(((size_t)NNZ_TOT * DIM + 255) / 256);
    const int ETILE_BLOCKS = (NNZ_TOT / 16 + 3) / 4;
    const int NTILE_BLOCKS = (N_CNT / 16 + 3) / 4;
    const int BD_BLOCKS    = (int)((BD + 255) / 256);

    hipMemsetAsync(scal, 0, 16 * sizeof(float), stream);
    concat_kernel<<<ND_BLOCKS, 256, 0, stream>>>(user_emb, item_emb, e0);

    // ---------------- base view: spmm + mask MLPs ----------------
    hipMemcpyAsync(eA, e0, ND * sizeof(float), hipMemcpyDeviceToDevice, stream);
    hipMemcpyAsync(vsum, e0, ND * sizeof(float), hipMemcpyDeviceToDevice, stream);
    {
        float* cur = eA; float* nxt = eB;
        for (int i = 0; i < LAYERS; ++i) {
            hipMemsetAsync(nxt, 0, ND * sizeof(float), stream);
            spmm_kernel<<<SPMM_BLOCKS, 256, 0, stream>>>(nxt, cur, adj_val, nullptr, adj_row, adj_col);
            edge_mlp_kernel<<<ETILE_BLOCKS, 128, 0, stream>>>(
                nxt, edge_w1 + (size_t)i * 2 * DIM * DIM, edge_b1 + i * DIM,
                edge_w2 + i * DIM, edge_b2 + i, edge_eps + (size_t)i * NNZ_TOT,
                adj_row, adj_col, emask + (size_t)i * NNZ_TOT);
            node_mlp_kernel<<<NTILE_BLOCKS, 128, 0, stream>>>(
                nxt, node_w1 + (size_t)i * DIM * DIM, node_b1 + i * DIM,
                node_w2 + i * DIM, node_b2 + i, node_eps + (size_t)i * N_CNT,
                nmask + (size_t)i * N_CNT);
            accum_kernel<<<ND_BLOCKS, 256, 0, stream>>>(vsum, nxt);
            float* t = cur; cur = nxt; nxt = t;
        }
    }
    gather_kernel<<<BD_BLOCKS, 256, 0, stream>>>(
        vsum, user_id, pos_item, neg_item, gbuf + 0 * BD, gbuf + 1 * BD, gbuf + 2 * BD);

    // ---------------- edge-masked view ----------------
    hipMemcpyAsync(eA, e0, ND * sizeof(float), hipMemcpyDeviceToDevice, stream);
    hipMemcpyAsync(vsum, e0, ND * sizeof(float), hipMemcpyDeviceToDevice, stream);
    {
        float* cur = eA; float* nxt = eB;
        for (int i = 0; i < LAYERS; ++i) {
            const float* mi = emask + (size_t)i * NNZ_TOT;
            reduce_mul_sum_kernel<<<512, 256, 0, stream>>>(
                adj_val, mi, NNZ_TOT, scal + 4, 1.0f / ((float)NNZ_HALF_C * LAYERS));
            hipMemsetAsync(nxt, 0, ND * sizeof(float), stream);
            spmm_kernel<<<SPMM_BLOCKS, 256, 0, stream>>>(nxt, cur, adj_val, mi, adj_row, adj_col);
            accum_kernel<<<ND_BLOCKS, 256, 0, stream>>>(vsum, nxt);
            float* t = cur; cur = nxt; nxt = t;
        }
    }
    gather_kernel<<<BD_BLOCKS, 256, 0, stream>>>(
        vsum, user_id, pos_item, neg_item, gbuf + 3 * BD, gbuf + 4 * BD, gbuf + 5 * BD);

    // ---------------- node-masked view ----------------
    hipMemcpyAsync(eA, e0, ND * sizeof(float), hipMemcpyDeviceToDevice, stream);
    hipMemcpyAsync(vsum, e0, ND * sizeof(float), hipMemcpyDeviceToDevice, stream);
    for (int i = 0; i < LAYERS; ++i) {
        const float* mi = nmask + (size_t)i * N_CNT;
        reduce_mul_sum_kernel<<<256, 256, 0, stream>>>(
            mi, nullptr, N_CNT, scal + 5, 1.0f / ((float)N_CNT * LAYERS));
        mix_kernel<<<ND_BLOCKS, 256, 0, stream>>>(eB, eA, walk, mi);
        hipMemsetAsync(eA, 0, ND * sizeof(float), stream);
        spmm_kernel<<<SPMM_BLOCKS, 256, 0, stream>>>(eA, eB, adj_val, nullptr, adj_row, adj_col);
        accum_kernel<<<ND_BLOCKS, 256, 0, stream>>>(vsum, eA);
    }
    gather_kernel<<<BD_BLOCKS, 256, 0, stream>>>(
        vsum, user_id, pos_item, neg_item, gbuf + 6 * BD, gbuf + 7 * BD, gbuf + 8 * BD);

    // ---------------- BPR losses (scal[0..2]) ----------------
    for (int v = 0; v < 3; ++v)
        bpr_kernel<<<(BATCH + 255) / 256, 256, 0, stream>>>(
            gbuf + (size_t)(3 * v + 0) * BD, gbuf + (size_t)(3 * v + 1) * BD,
            gbuf + (size_t)(3 * v + 2) * BD, scal + v);

    // ---------------- SSL: normalize u/p rows in place, pre-convert, InfoNCE
    for (int v = 0; v < 3; ++v) {
        normalize_kernel<<<(BATCH + 255) / 256, 256, 0, stream>>>(gbuf + (size_t)(3 * v + 0) * BD);
        normalize_kernel<<<(BATCH + 255) / 256, 256, 0, stream>>>(gbuf + (size_t)(3 * v + 1) * BD);
    }
    frag_b_convert<<<BD_BLOCKS, 256, 0, stream>>>(gbuf + 0 * BD, fr_bu);
    frag_b_convert<<<BD_BLOCKS, 256, 0, stream>>>(gbuf + 1 * BD, fr_bi);
    frag_a_convert<<<BD_BLOCKS, 256, 0, stream>>>(gbuf + 3 * BD, fr_eu);
    frag_a_convert<<<BD_BLOCKS, 256, 0, stream>>>(gbuf + 6 * BD, fr_nu);
    frag_a_convert<<<BD_BLOCKS, 256, 0, stream>>>(gbuf + 4 * BD, fr_ei);
    frag_a_convert<<<BD_BLOCKS, 256, 0, stream>>>(gbuf + 7 * BD, fr_ni);

    // ssl(eu,bu) + ssl(nu,bu) + ssl(ei,bi) + ssl(ni,bi) -> scal[3]
    ssl_kernel<<<BATCH / 16 / 4, 128, 0, stream>>>(fr_eu, fr_bu, gbuf + 3 * BD, gbuf + 0 * BD, scal + 3);
    ssl_kernel<<<BATCH / 16 / 4, 128, 0, stream>>>(fr_nu, fr_bu, gbuf + 6 * BD, gbuf + 0 * BD, scal + 3);
    ssl_kernel<<<BATCH / 16 / 4, 128, 0, stream>>>(fr_ei, fr_bi, gbuf + 4 * BD, gbuf + 1 * BD, scal + 3);
    ssl_kernel<<<BATCH / 16 / 4, 128, 0, stream>>>(fr_ni, fr_bi, gbuf + 7 * BD, gbuf + 1 * BD, scal + 3);

    finalize_kernel<<<1, 1, 0, stream>>>(scal, (float*)d_out);
}
